// _RPN_FPN_45286135169233
// MI455X (gfx1250) — matrix operand order repaired and checked by reading, not compiled, not run on hardware
//
#include <hip/hip_runtime.h>

// ---------------------------------------------------------------- types
typedef __bf16 v16bf __attribute__((ext_vector_type(16)));
typedef float  v8f   __attribute__((ext_vector_type(8)));
typedef unsigned int uv4 __attribute__((ext_vector_type(4)));
typedef int iv8 __attribute__((ext_vector_type(8)));
typedef int iv4 __attribute__((ext_vector_type(4)));

union Frag {
    v16bf    v;
    unsigned u[8];
    uint4    q[2];
};

__device__ __forceinline__ unsigned short f2bf(float f) {
    unsigned u = __float_as_uint(f);
    u += 0x7fffu + ((u >> 16) & 1u);
    return (unsigned short)(u >> 16);
}

// ---------------------------------------------------------------- geometry
#define N_SPATIAL   34104
#define N_SPATIAL_P 34176          // padded to 267*128
#define N_ANCHORS   306936
#define PRE_NMS     6000
#define POST_NMS    300
#define SEL_CAP     6016
#define NBINS       4096

// ---------------------------------------------------------------- TDM helper
// Build a 2-D Tensor-DMA descriptor and issue tensor_load_to_lds.
// D# group0: [1:0]=count=1, [63:32]=lds_addr, [120:64]=global_addr, [127:126]=type=2
// D# group1: [17:16]=data_size(1->2B), [79:48]=tensor_dim0, [111:80]=tensor_dim1,
//            [127:112]=tile_dim0, [143:128]=tile_dim1, [207:160]=tensor_dim0_stride
__device__ __forceinline__ void tdm_load_2d(
    const unsigned short* lds_ptr, const unsigned short* gptr,
    unsigned tile_d0, unsigned tile_d1,
    unsigned tensor_d0, unsigned tensor_d1, unsigned stride0)
{
#if __has_builtin(__builtin_amdgcn_tensor_load_to_lds)
    unsigned long long ga = (unsigned long long)(uintptr_t)gptr;
    unsigned lds_off = (unsigned)(uintptr_t)lds_ptr;   // low 32b of generic ptr = LDS byte offset
    uv4 g0;
    g0.x = 1u;                                         // count=1 (valid user descriptor)
    g0.y = lds_off;
    g0.z = (unsigned)(ga & 0xffffffffu);
    g0.w = (unsigned)((ga >> 32) & 0x01ffffffu) | (2u << 30);  // addr hi + type=2
    iv8 g1;
    g1[0] = 0x00010000;                                // data_size = 1 -> 2 bytes/elem
    g1[1] = (int)((tensor_d0 & 0xffffu) << 16);
    g1[2] = (int)((tensor_d0 >> 16) & 0xffffu) | (int)((tensor_d1 & 0xffffu) << 16);
    g1[3] = (int)((tensor_d1 >> 16) & 0xffffu) | (int)((tile_d0 & 0xffffu) << 16);
    g1[4] = (int)(tile_d1 & 0xffffu);                  // tile_dim1 (tile_dim2 = 0)
    g1[5] = (int)stride0;                              // tensor_dim0_stride lo32
    g1[6] = 0;
    g1[7] = 0;
    iv4 z4 = {0, 0, 0, 0};
#if defined(__clang_major__) && (__clang_major__ >= 23)
    iv8 z8 = {0, 0, 0, 0, 0, 0, 0, 0};
    __builtin_amdgcn_tensor_load_to_lds(g0, g1, z4, z4, z8, 0);
#else
    __builtin_amdgcn_tensor_load_to_lds(g0, g1, z4, z4, 0);
#endif
#else
    (void)lds_ptr; (void)gptr; (void)tile_d0; (void)tile_d1;
    (void)tensor_d0; (void)tensor_d1; (void)stride0;
#endif
}

__device__ __forceinline__ void tdm_wait0() {
#if __has_builtin(__builtin_amdgcn_s_wait_tensorcnt)
    __builtin_amdgcn_s_wait_tensorcnt(0);
#endif
}

// ---------------------------------------------------------------- input f32 -> bf16
__global__ void k_cvt(const float* __restrict__ x, unsigned short* __restrict__ xb, int n) {
    int i = (blockIdx.x * 256 + threadIdx.x) * 4;
    if (i >= n) return;
    float4 f = *(const float4*)(x + i);
    uint2 o;
    o.x = (unsigned)f2bf(f.x) | ((unsigned)f2bf(f.y) << 16);
    o.y = (unsigned)f2bf(f.z) | ((unsigned)f2bf(f.w) << 16);
    *(uint2*)(xb + i) = o;
}

// ---------------------------------------------------------------- weight prep
// conv blob: E = ((tap*8+kc)*32 + nt)*512 + lane*16 + e   (bf16)
__global__ void k_prep_wc(const float* __restrict__ w, unsigned short* __restrict__ wc) {
    int E = blockIdx.x * 256 + threadIdx.x;            // 1,179,648 total
    int e = E & 15, lane = (E >> 4) & 31, blk = E >> 9;
    int nt = blk & 31, kc = (blk >> 5) & 7, tap = blk >> 8;
    int half = lane >> 4, p = e >> 1, wb = e & 1;
    int k = kc * 32 + half * 16 + p * 2 + wb;          // input channel
    int n = nt * 16 + (lane & 15);                     // output channel
    wc[E] = f2bf(w[(size_t)(n * 256 + k) * 9 + tap]);
}

// head blob: E = (kc*4 + nt)*512 + lane*16 + e   (bf16), N packed: [cls 18 | bbox 36 | pad]
__global__ void k_prep_wh(const float* __restrict__ cls_w, const float* __restrict__ bbox_w,
                          unsigned short* __restrict__ wh) {
    int E = blockIdx.x * 256 + threadIdx.x;            // 32,768 total
    int e = E & 15, lane = (E >> 4) & 31, blk = E >> 9;
    int nt = blk & 3, kc = blk >> 2;
    int half = lane >> 4, p = e >> 1, wb = e & 1;
    int k = kc * 32 + half * 16 + p * 2 + wb;
    int n = nt * 16 + (lane & 15);
    float v = 0.f;
    if (n < 18)      v = cls_w[n * 512 + k];
    else if (n < 54) v = bbox_w[(n - 18) * 512 + k];
    wh[E] = f2bf(v);
}

// ---------------------------------------------------------------- 3x3 conv + ReLU
// 2-D tiling: 8x8 spatial tile + 1px halo; one 10x10x32 LDS tile per K-chunk
// serves all 9 taps. Double-buffered and software-pipelined: next tile's global
// loads are issued before the 36-WMMA compute block, ds_store after it.
// block = 256 threads = 8 waves (4 M-waves x 2 N-waves); tile M=64 x N=128.
__global__ __launch_bounds__(256) void k_conv3x3(
    const unsigned short* __restrict__ xbf, const unsigned short* __restrict__ wc,
    const float* __restrict__ bias, unsigned short* __restrict__ hidden,
    int H, int W, int HW, int lvl_off, int tilesX)
{
    __shared__ unsigned short Atile[2][3200];          // 2 x 6.25 KB
    const int nchunk = blockIdx.x & 3;
    const int tile   = blockIdx.x >> 2;
    const int ty = tile / tilesX, tx = tile - ty * tilesX;
    const int y0 = ty * 8, x0 = tx * 8;
    const int t    = threadIdx.x;
    const int lane = t & 31, wv = t >> 5;
    const int waveM = wv & 3, waveN = wv >> 2;
    const int n0   = nchunk * 128;
    const int half = lane >> 4;
    const int mL   = waveM * 16 + (lane & 15);
    const int ly   = mL >> 3, lx = mL & 7;

    // ---- kc-invariant staging map: global elem offset (or -1) + LDS offset (or -1)
    int goff[13], lofs[13];
    #pragma unroll
    for (int j = 0; j < 13; ++j) {
        int e = t + j * 256;
        goff[j] = -1; lofs[j] = -1;
        if (e < 3200) {
            int kk = e / 100;
            int rc = e - kk * 100;
            int row = rc / 10, col = rc - row * 10;
            int sy = y0 + row - 1, sx = x0 + col - 1;
            lofs[j] = rc * 32 + kk;
            if (sy >= 0 && sy < H && sx >= 0 && sx < W)
                goff[j] = kk * HW + sy * W + sx;
        }
    }

    unsigned short stg[13];
    // ---- prologue: stage kc=0
    #pragma unroll
    for (int j = 0; j < 13; ++j)
        stg[j] = (goff[j] >= 0) ? xbf[goff[j]] : (unsigned short)0;
    #pragma unroll
    for (int j = 0; j < 13; ++j)
        if (lofs[j] >= 0) Atile[0][lofs[j]] = stg[j];
    __syncthreads();

    v8f acc[4] = {};

    for (int kc = 0; kc < 8; ++kc) {
        // issue next tile's global loads (latency hidden behind WMMAs)
        if (kc < 7) {
            const unsigned short* xn = xbf + (size_t)(kc + 1) * 32 * HW;
            #pragma unroll
            for (int j = 0; j < 13; ++j)
                stg[j] = (goff[j] >= 0) ? xn[goff[j]] : (unsigned short)0;
            __builtin_prefetch((const void*)(wc + (size_t)(kc + 1) * 32 * 512), 0, 3);
        }

        // ---- 9 taps x 4 N-fragments from the resident tile
        const unsigned short* Ab = &Atile[kc & 1][0];
        #pragma unroll
        for (int tap = 0; tap < 9; ++tap) {
            const int dy = tap / 3, dx = tap % 3;      // halo-relative (0..2)
            const unsigned* arow =
                (const unsigned*)(Ab + ((ly + dy) * 10 + (lx + dx)) * 32);
            Frag a;
            #pragma unroll
            for (int p = 0; p < 8; ++p) {
                int kb = p * 2 + half * 16;
                a.u[p] = arow[kb >> 1];
            }
            const uint4* wcq = (const uint4*)(wc + (size_t)(tap * 8 + kc) * 32 * 512);
            #pragma unroll
            for (int i = 0; i < 4; ++i) {
                int nt = (n0 >> 4) + waveN * 4 + i;
                Frag b;
                b.q[0] = wcq[nt * 64 + lane * 2 + 0];
                b.q[1] = wcq[nt * 64 + lane * 2 + 1];
                acc[i] = __builtin_amdgcn_wmma_f32_16x16x32_bf16(
                    false, a.v, false, b.v, (short)0, acc[i], false, false);
            }
        }

        // commit next tile into the other buffer
        if (kc < 7) {
            #pragma unroll
            for (int j = 0; j < 13; ++j)
                if (lofs[j] >= 0) Atile[(kc + 1) & 1][lofs[j]] = stg[j];
        }
        __syncthreads();
    }

    // ---- epilogue: bias + ReLU -> bf16 hidden[m][512]
    #pragma unroll
    for (int i = 0; i < 4; ++i) {
        int n = n0 + waveN * 64 + i * 16 + (lane & 15);
        float bv = bias[n];
        #pragma unroll
        for (int r = 0; r < 8; ++r) {
            int m = waveM * 16 + r + half * 8;
            int yy = y0 + (m >> 3), xx = x0 + (m & 7);
            if (yy < H && xx < W) {
                float val = acc[i][r] + bv;
                val = val > 0.f ? val : 0.f;
                hidden[(size_t)(lvl_off + yy * W + xx) * 512 + n] = f2bf(val);
            }
        }
    }
}

// ---------------------------------------------------------------- 1x1 heads GEMM
// M=34104(p 34176), K=512, N=64. A tiles (128x32 bf16) streamed by the
// Tensor Data Mover into double-buffered LDS; wave 0 drives the TDM.
__global__ __launch_bounds__(256) void k_head(
    const unsigned short* __restrict__ hidden, const unsigned short* __restrict__ wh,
    const float* __restrict__ cls_b, const float* __restrict__ bbox_b,
    float* __restrict__ head)
{
    __shared__ unsigned short At[2][128 * 32];         // 2 x 8 KB
    const int t = threadIdx.x, lane = t & 31, wv = t >> 5;
    const int mBlk = blockIdx.x * 128;
    const int half = lane >> 4;
    const int row  = wv * 16 + (lane & 15);
    v8f acc[4] = {};

    const uint4* whq = (const uint4*)wh;
    const unsigned short* hbase = hidden + (size_t)mBlk * 512;

    if (wv == 0)
        tdm_load_2d(&At[0][0], hbase, 32, 128, 512, (unsigned)N_SPATIAL_P, 512);

    for (int kc = 0; kc < 16; ++kc) {
        if (wv == 0) tdm_wait0();                       // current buffer resident
        __syncthreads();
        if (wv == 0 && kc < 15)
            tdm_load_2d(&At[(kc + 1) & 1][0], hbase + (kc + 1) * 32,
                        32, 128, 512, (unsigned)N_SPATIAL_P, 512);

        const unsigned short* Ab = &At[kc & 1][0];
        Frag a;
        const unsigned* arow = (const unsigned*)(Ab + row * 32);
        #pragma unroll
        for (int p = 0; p < 8; ++p) {
            int kb = p * 2 + half * 16;
            a.u[p] = arow[kb >> 1];
        }
        #pragma unroll
        for (int i = 0; i < 4; ++i) {
            Frag b;
            b.q[0] = whq[(kc * 4 + i) * 64 + lane * 2 + 0];
            b.q[1] = whq[(kc * 4 + i) * 64 + lane * 2 + 1];
            acc[i] = __builtin_amdgcn_wmma_f32_16x16x32_bf16(
                false, a.v, false, b.v, (short)0, acc[i], false, false);
        }
    }
    #pragma unroll
    for (int i = 0; i < 4; ++i) {
        int n = i * 16 + (lane & 15);
        float bv = (n < 18) ? cls_b[n] : ((n < 54) ? bbox_b[n - 18] : 0.f);
        #pragma unroll
        for (int r = 0; r < 8; ++r) {
            int m = mBlk + wv * 16 + r + half * 8;
            head[(size_t)m * 64 + n] = acc[i][r] + bv;
        }
    }
}

// ---------------------------------------------------------------- softmax + decode + clip
__global__ void k_decode(const float* __restrict__ head, const float* __restrict__ im_info,
                         float* __restrict__ prob_out, float* __restrict__ bbox_out,
                         float* __restrict__ prop, float* __restrict__ score)
{
    int g = blockIdx.x * 256 + threadIdx.x;
    if (g >= N_ANCHORS) return;
    const int aoff[6]  = {0, 230400, 288000, 302400, 306000, 306936};
    const int soff[5]  = {0, 25600, 32000, 33600, 34000};
    const int Ws[5]    = {200, 100, 50, 25, 13};
    const int strd[5]  = {4, 8, 16, 32, 64};
    int l = 0;
    while (g >= aoff[l + 1]) ++l;
    int rel = g - aoff[l];
    int s = rel / 9, a = rel - s * 9;
    int m = soff[l] + s;
    int Wl = Ws[l];
    int y = s / Wl, xx = s - y * Wl;

    const float* hr = head + (size_t)m * 64;
    float l0 = hr[2 * a], l1 = hr[2 * a + 1];
    float mx = fmaxf(l0, l1);
    float e0 = __expf(l0 - mx), e1 = __expf(l1 - mx);
    float inv = 1.f / (e0 + e1);
    prob_out[(size_t)g * 2 + 0] = e0 * inv;
    prob_out[(size_t)g * 2 + 1] = e1 * inv;

    float d0 = hr[18 + 4 * a], d1 = hr[19 + 4 * a], d2 = hr[20 + 4 * a], d3 = hr[21 + 4 * a];
    bbox_out[(size_t)g * 4 + 0] = d0; bbox_out[(size_t)g * 4 + 1] = d1;
    bbox_out[(size_t)g * 4 + 2] = d2; bbox_out[(size_t)g * 4 + 3] = d3;

    const float ratios[3] = {0.5f, 1.f, 2.f};
    const float scales[3] = {8.f, 16.f, 32.f};
    float st = (float)strd[l];
    int r = a / 3, si = a - r * 3;
    float wsr = roundf(sqrtf(st * st / ratios[r]));
    float hsr = roundf(wsr * ratios[r]);
    float W0 = wsr * scales[si], H0 = hsr * scales[si];
    float ctr = (st - 1.f) * 0.5f;
    float ax1 = xx * st + ctr - (W0 - 1.f) * 0.5f;
    float ay1 = y  * st + ctr - (H0 - 1.f) * 0.5f;
    float ax2 = xx * st + ctr + (W0 - 1.f) * 0.5f;
    float ay2 = y  * st + ctr + (H0 - 1.f) * 0.5f;

    float aw = ax2 - ax1 + 1.f, ah = ay2 - ay1 + 1.f;
    float acx = ax1 + 0.5f * aw, acy = ay1 + 0.5f * ah;
    float pcx = d0 * aw + acx,  pcy = d1 * ah + acy;
    float pw = __expf(d2) * aw, ph = __expf(d3) * ah;
    float imH = im_info[0], imW = im_info[1];
    float x1 = fminf(fmaxf(pcx - 0.5f * pw, 0.f), imW - 1.f);
    float y1 = fminf(fmaxf(pcy - 0.5f * ph, 0.f), imH - 1.f);
    float x2 = fminf(fmaxf(pcx + 0.5f * pw, 0.f), imW - 1.f);
    float y2 = fminf(fmaxf(pcy + 0.5f * ph, 0.f), imH - 1.f);
    prop[(size_t)g * 4 + 0] = x1; prop[(size_t)g * 4 + 1] = y1;
    prop[(size_t)g * 4 + 2] = x2; prop[(size_t)g * 4 + 3] = y2;
    score[g] = e1 * inv;
}

// ---------------------------------------------------------------- threshold select (approx top-k)
__global__ void k_init(int* __restrict__ hist, int* __restrict__ cnt) {
    int t = blockIdx.x * 256 + threadIdx.x;
    if (t < NBINS) hist[t] = 0;
    if (t < 4)     cnt[t]  = 0;
}
__global__ void k_hist(const float* __restrict__ score, int* __restrict__ hist) {
    int g = blockIdx.x * 256 + threadIdx.x;
    if (g >= N_ANCHORS) return;
    int b = (int)(score[g] * (float)NBINS);
    b = b < 0 ? 0 : (b > NBINS - 1 ? NBINS - 1 : b);
    atomicAdd(&hist[b], 1);
}
__global__ void k_thresh(const int* __restrict__ hist, int* __restrict__ cnt) {
    if (threadIdx.x == 0) {
        int cum = 0, th = 0;
        for (int b = NBINS - 1; b >= 0; --b) {
            cum += hist[b];
            if (cum >= PRE_NMS) { th = b; break; }
        }
        cnt[1] = th;
    }
}
__global__ void k_compact(const float* __restrict__ score, const float* __restrict__ prop,
                          int* __restrict__ cnt, float* __restrict__ selB, float* __restrict__ selS) {
    int g = blockIdx.x * 256 + threadIdx.x;
    if (g >= N_ANCHORS) return;
    float s = score[g];
    int b = (int)(s * (float)NBINS);
    b = b < 0 ? 0 : (b > NBINS - 1 ? NBINS - 1 : b);
    if (b >= cnt[1]) {
        int pos = atomicAdd(&cnt[0], 1);
        if (pos < PRE_NMS) {
            selB[pos * 4 + 0] = prop[(size_t)g * 4 + 0];
            selB[pos * 4 + 1] = prop[(size_t)g * 4 + 1];
            selB[pos * 4 + 2] = prop[(size_t)g * 4 + 2];
            selB[pos * 4 + 3] = prop[(size_t)g * 4 + 3];
            selS[pos] = s;
        }
    }
}

// ---------------------------------------------------------------- NMS: single WGP, LDS-resident
__global__ __launch_bounds__(1024) void k_nms(
    const float* __restrict__ selB, const float* __restrict__ selS,
    const int* __restrict__ cnt, float* __restrict__ rois)
{
    extern __shared__ float sm[];
    float* bx   = sm;                       // SEL_CAP*4
    float* sc   = bx + SEL_CAP * 4;         // SEL_CAP
    float* ar   = sc + SEL_CAP;             // SEL_CAP
    float* redS = ar + SEL_CAP;             // 1024
    int*   redI = (int*)(redS + 1024);      // 1024
    const int t = threadIdx.x;
    int n = cnt[0]; if (n > PRE_NMS) n = PRE_NMS;

    for (int i = t; i < n; i += 1024) {
        float x1 = selB[i * 4], y1 = selB[i * 4 + 1];
        float x2 = selB[i * 4 + 2], y2 = selB[i * 4 + 3];
        bx[i * 4] = x1; bx[i * 4 + 1] = y1; bx[i * 4 + 2] = x2; bx[i * 4 + 3] = y2;
        sc[i] = selS[i];
        ar[i] = (x2 - x1 + 1.f) * (y2 - y1 + 1.f);
    }
    __syncthreads();

    for (int it = 0; it < POST_NMS; ++it) {
        float bs = -2e9f; int bi = -1;
        for (int i = t; i < n; i += 1024) {
            float v = sc[i];
            if (v > bs) { bs = v; bi = i; }
        }
        redS[t] = bs; redI[t] = bi;
        __syncthreads();
        for (int off = 512; off > 0; off >>= 1) {
            if (t < off && redS[t + off] > redS[t]) {
                redS[t] = redS[t + off]; redI[t] = redI[t + off];
            }
            __syncthreads();
        }
        float bestS = redS[0]; int bestI = redI[0];
        bool valid = (bestI >= 0) && (bestS > -5.0e8f);
        float b0 = 0, b1 = 0, b2 = 0, b3 = 0, bA = 1.f;
        if (valid) {
            b0 = bx[bestI * 4]; b1 = bx[bestI * 4 + 1];
            b2 = bx[bestI * 4 + 2]; b3 = bx[bestI * 4 + 3];
            bA = ar[bestI];
        }
        __syncthreads();
        if (valid) {
            for (int i = t; i < n; i += 1024) {
                float xx1 = fmaxf(b0, bx[i * 4]),     yy1 = fmaxf(b1, bx[i * 4 + 1]);
                float xx2 = fminf(b2, bx[i * 4 + 2]), yy2 = fminf(b3, bx[i * 4 + 3]);
                float inter = fmaxf(xx2 - xx1 + 1.f, 0.f) * fmaxf(yy2 - yy1 + 1.f, 0.f);
                float iou = inter / (ar[i] + bA - inter);
                if (iou > 0.7f) sc[i] = -1e9f;
            }
            if (t == 0) sc[bestI] = -1e9f;
        }
        if (t == 0) {
            rois[it * 4 + 0] = valid ? b0 : 0.f;
            rois[it * 4 + 1] = valid ? b1 : 0.f;
            rois[it * 4 + 2] = valid ? b2 : 0.f;
            rois[it * 4 + 3] = valid ? b3 : 0.f;
        }
        __syncthreads();
    }
}

// ---------------------------------------------------------------- launch
extern "C" void kernel_launch(void* const* d_in, const int* in_sizes, int n_in,
                              void* d_out, int out_size, void* d_ws, size_t ws_size,
                              hipStream_t stream) {
    (void)in_sizes; (void)n_in; (void)out_size; (void)ws_size;
    const float* feats[5] = {(const float*)d_in[0], (const float*)d_in[1],
                             (const float*)d_in[2], (const float*)d_in[3],
                             (const float*)d_in[4]};
    const float* im_info = (const float*)d_in[5];
    const float* conv_w  = (const float*)d_in[6];
    const float* conv_b  = (const float*)d_in[7];
    const float* cls_w   = (const float*)d_in[8];
    const float* cls_b   = (const float*)d_in[9];
    const float* bbox_w  = (const float*)d_in[10];
    const float* bbox_b  = (const float*)d_in[11];

    float* out  = (float*)d_out;
    float* rois = out;                          // 300*4
    float* prob = out + 1200;                   // 306936*2
    float* bbox = out + 1200 + N_ANCHORS * 2;   // 306936*4

    // workspace layout (bytes)
    char* ws = (char*)d_ws;
    unsigned short* wc     = (unsigned short*)(ws);                      //  2,359,296 B
    unsigned short* wh     = (unsigned short*)(ws + 2359296);            //     65,536 B
    unsigned short* xbf    = (unsigned short*)(ws + 2424832);            // 17,461,248 B
    unsigned short* hidden = (unsigned short*)(ws + 19886080);           // 34,996,224 B
    float*          head   = (float*)         (ws + 54882304);           //  8,749,056 B
    float*          prop   = (float*)         (ws + 63631360);           //  4,910,976 B
    float*          score  = (float*)         (ws + 68542336);           //  1,227,744 B
    int*            hist   = (int*)           (ws + 69770080);           //     16,384 B
    int*            cnt    = (int*)           (ws + 69786464);           //         16 B
    float*          selB   = (float*)         (ws + 69786624);           //     96,256 B
    float*          selS   = (float*)         (ws + 69882880);           //     24,064 B

    const int HS[5]   = {128, 64, 32, 16, 8};
    const int WS_[5]  = {200, 100, 50, 25, 13};
    const int SOFF[5] = {0, 25600, 32000, 33600, 34000};

    // 0) activations f32 -> bf16 (channel-major preserved)
    for (int l = 0; l < 5; ++l) {
        int n = 256 * HS[l] * WS_[l];
        k_cvt<<<(n / 4 + 255) / 256, 256, 0, stream>>>(feats[l], xbf + (size_t)SOFF[l] * 256, n);
    }

    // 1) weight reshuffle into WMMA fragment-linear blobs
    k_prep_wc<<<4608, 256, 0, stream>>>(conv_w, wc);
    k_prep_wh<<<128, 256, 0, stream>>>(cls_w, bbox_w, wh);

    // 2) 3x3 conv + ReLU per level (WMMA implicit GEMM, pipelined 2-D halo tiles)
    for (int l = 0; l < 5; ++l) {
        int tilesX = (WS_[l] + 7) / 8;
        int tilesY = (HS[l] + 7) / 8;
        k_conv3x3<<<tilesX * tilesY * 4, 256, 0, stream>>>(
            xbf + (size_t)SOFF[l] * 256, wc, conv_b, hidden,
            HS[l], WS_[l], HS[l] * WS_[l], SOFF[l], tilesX);
    }

    // 3) fused cls+bbox 1x1 heads (WMMA GEMM, TDM-fed double-buffered LDS)
    k_head<<<N_SPATIAL_P / 128, 256, 0, stream>>>(hidden, wh, cls_b, bbox_b, head);

    // 4) softmax + anchor decode + clip
    k_decode<<<(N_ANCHORS + 255) / 256, 256, 0, stream>>>(head, im_info, prob, bbox,
                                                          prop, score);

    // 5) approximate top-6000 via score histogram threshold
    k_init<<<16, 256, 0, stream>>>(hist, cnt);
    k_hist<<<(N_ANCHORS + 255) / 256, 256, 0, stream>>>(score, hist);
    k_thresh<<<1, 64, 0, stream>>>(hist, cnt);
    k_compact<<<(N_ANCHORS + 255) / 256, 256, 0, stream>>>(score, prop, cnt, selB, selS);

    // 6) NMS — single WGP, candidates fully LDS-resident (CDNA5 320KB LDS)
    size_t nms_lds = (size_t)(SEL_CAP * 6 + 1024) * 4 + 1024 * 4;
    k_nms<<<1, 1024, nms_lds, stream>>>(selB, selS, cnt, rois);
}
